// GNN_7713761264053
// MI455X (gfx1250) — matrix operand order repaired and checked by reading, not compiled, not run on hardware
//
#include <hip/hip_runtime.h>

#define N_NODES 50000
#define N_EDGES 600000
#define LAT 128
#define EDGE_TILES (N_EDGES / 16)   // 37500
#define NODE_TILES (N_NODES / 16)   // 3125

typedef __attribute__((ext_vector_type(16))) _Float16 v16h;
typedef __attribute__((ext_vector_type(8)))  _Float16 v8h;
typedef __attribute__((ext_vector_type(8)))  float    v8f;

__device__ __forceinline__ v8f zero_v8f() {
  v8f z;
#pragma unroll
  for (int j = 0; j < 8; ++j) z[j] = 0.0f;
  return z;
}

__device__ __forceinline__ v16h combine16(v8h lo, v8h hh) {
  v16h a;
#pragma unroll
  for (int i = 0; i < 8; ++i) { a[i] = lo[i]; a[i + 8] = hh[i]; }
  return a;
}

// ---------------- encoders (tiny K: scalar) ----------------
__global__ __launch_bounds__(256) void node_enc(const float* __restrict__ nf,
                                                const float* __restrict__ W,
                                                const float* __restrict__ b,
                                                _Float16* __restrict__ h16) {
  long idx = (long)blockIdx.x * 256 + threadIdx.x;       // over N_NODES*LAT
  int col = (int)(idx & (LAT - 1));
  long row = idx >> 7;
  float s = b[col];
#pragma unroll
  for (int k = 0; k < 8; ++k) s += nf[row * 8 + k] * W[k * LAT + col];
  h16[idx] = (_Float16)s;
}

__global__ __launch_bounds__(256) void edge_enc(const float* __restrict__ ef,
                                                const float* __restrict__ W,
                                                const float* __restrict__ b,
                                                _Float16* __restrict__ e16) {
  long idx = (long)blockIdx.x * 256 + threadIdx.x;       // over N_EDGES*LAT
  int col = (int)(idx & (LAT - 1));
  long row = idx >> 7;
  float s = b[col];
#pragma unroll
  for (int k = 0; k < 4; ++k) s += ef[row * 4 + k] * W[k * LAT + col];
  e16[idx] = (_Float16)s;
}

// ---------------- weight pack: f32 row-major -> f16 WMMA-B layout ----------------
// B layout (wave32, 16x16x32 f16): element i (0..15) of lane L holds
//   K = kstep*32 + (i&7) + 8*(L>>4) + 16*(i>>3),  N = ntile*16 + (L&15)
__global__ __launch_bounds__(256) void pack_w(const float* __restrict__ We,
                                              const float* __restrict__ Wn,
                                              _Float16* __restrict__ We16,
                                              _Float16* __restrict__ Wn16) {
  int idx = blockIdx.x * 256 + threadIdx.x;
  const int EDGE_TOT = 3 * 12 * 8 * 32 * 16;  // 147456
  const int NODE_TOT = 3 * 8 * 8 * 32 * 16;   // 98304
  if (idx < EDGE_TOT) {
    int l = idx / 49152, r = idx % 49152;
    int ks = r / 4096, r2 = r % 4096;
    int nt = r2 / 512, r3 = r2 % 512;
    int lane = r3 / 16, i = r3 % 16;
    int K = ks * 32 + (i & 7) + 8 * (lane >> 4) + 16 * (i >> 3);
    int N = nt * 16 + (lane & 15);
    We16[idx] = (_Float16)We[((long)l * 384 + K) * LAT + N];
  } else if (idx < EDGE_TOT + NODE_TOT) {
    int j = idx - EDGE_TOT;
    int l = j / 32768, r = j % 32768;
    int ks = r / 4096, r2 = r % 4096;
    int nt = r2 / 512, r3 = r2 % 512;
    int lane = r3 / 16, i = r3 % 16;
    int K = ks * 32 + (i & 7) + 8 * (lane >> 4) + 16 * (i >> 3);
    int N = nt * 16 + (lane & 15);
    Wn16[j] = (_Float16)Wn[((long)l * 256 + K) * LAT + N];
  }
}

__global__ __launch_bounds__(256) void zero_f32(float* __restrict__ p, long n) {
  long idx = (long)blockIdx.x * 256 + threadIdx.x;
  if (idx < n) p[idx] = 0.0f;
}

// ---------------- edge MLP: e = relu([h_src|h_dst|e] @ We + be); agg += scatter ----
__global__ __launch_bounds__(256) void edge_mlp(const _Float16* __restrict__ h16,
                                                _Float16* __restrict__ e16,
                                                float* __restrict__ e_out,
                                                const int* __restrict__ src,
                                                const int* __restrict__ dst,
                                                const _Float16* __restrict__ We16,
                                                const float* __restrict__ be,
                                                float* __restrict__ agg,
                                                int layer, int write_out) {
  __shared__ __align__(32) _Float16 Bs[12 * 8 * 32 * 16];  // 96 KB packed W_edge[l]
  {
    const uint4* gs = (const uint4*)(We16 + (long)layer * 49152);
    uint4* gd = (uint4*)Bs;
    for (int i = threadIdx.x; i < 49152 * 2 / 16; i += 256) gd[i] = gs[i];
  }
  __syncthreads();

  int wave = threadIdx.x >> 5;
  int lane = threadIdx.x & 31;
  int m = lane & 15, hi = lane >> 4;
  int tile = blockIdx.x * 8 + wave;
  if (tile >= EDGE_TILES) return;  // uniform per wave: EXEC stays all-1s
  int e0 = tile * 16;
  int eid = e0 + m;
  int srow = src[eid], drow = dst[eid];

  v8f acc[8];
#pragma unroll
  for (int n = 0; n < 8; ++n) acc[n] = zero_v8f();

#pragma unroll
  for (int k = 0; k < 12; ++k) {
    const _Float16* rowp;
    if (k < 4)      rowp = h16 + (long)srow * LAT;
    else if (k < 8) rowp = h16 + (long)drow * LAT;
    else            rowp = e16 + (long)eid * LAT;
    int cb = (k & 3) * 32;
    v8h lo = *(const v8h*)(rowp + cb + 8 * hi);
    v8h hh = *(const v8h*)(rowp + cb + 16 + 8 * hi);
    v16h a = combine16(lo, hh);
#pragma unroll
    for (int n = 0; n < 8; ++n) {
      v16h b = *(const v16h*)(Bs + ((k * 8 + n) * 32 + lane) * 16);
      acc[n] = __builtin_amdgcn_wmma_f32_16x16x32_f16(
          false, a, false, b, (short)0, acc[n], false, false);
    }
  }

  const float* bep = be + layer * LAT;
#pragma unroll
  for (int n = 0; n < 8; ++n) {
    float bias = bep[n * 16 + m];
    int col = n * 16 + m;
#pragma unroll
    for (int v = 0; v < 8; ++v) {
      float x = acc[n][v] + bias;
      x = x > 0.0f ? x : 0.0f;
      int row = e0 + v + 8 * hi;            // C layout: M = v + 8*hi, N = col
      if (write_out) e_out[(long)row * LAT + col] = x;
      e16[(long)row * LAT + col] = (_Float16)x;
      atomicAdd(&agg[(long)dst[row] * LAT + col], x);
    }
  }
}

// ---------------- node MLP: h = relu([h|agg] @ Wn + bn) ----------------
__global__ __launch_bounds__(256) void node_mlp(const float* __restrict__ agg,
                                                _Float16* __restrict__ h16,
                                                float* __restrict__ h_out,
                                                const _Float16* __restrict__ Wn16,
                                                const float* __restrict__ bn,
                                                int layer, int write_out) {
  __shared__ __align__(32) _Float16 Bs[8 * 8 * 32 * 16];  // 64 KB packed W_node[l]
  {
    const uint4* gs = (const uint4*)(Wn16 + (long)layer * 32768);
    uint4* gd = (uint4*)Bs;
    for (int i = threadIdx.x; i < 32768 * 2 / 16; i += 256) gd[i] = gs[i];
  }
  __syncthreads();

  int wave = threadIdx.x >> 5;
  int lane = threadIdx.x & 31;
  int m = lane & 15, hi = lane >> 4;
  int tile = blockIdx.x * 8 + wave;
  if (tile >= NODE_TILES) return;
  int n0 = tile * 16;
  int nid = n0 + m;

  v8f acc[8];
#pragma unroll
  for (int n = 0; n < 8; ++n) acc[n] = zero_v8f();

#pragma unroll
  for (int k = 0; k < 8; ++k) {
    v16h a;
    if (k < 4) {
      const _Float16* rp = h16 + (long)nid * LAT + k * 32;
      v8h lo = *(const v8h*)(rp + 8 * hi);
      v8h hh = *(const v8h*)(rp + 16 + 8 * hi);
      a = combine16(lo, hh);
    } else {
      const float* rp = agg + (long)nid * LAT + (k - 4) * 32;
      float4 f0 = *(const float4*)(rp + 8 * hi);
      float4 f1 = *(const float4*)(rp + 8 * hi + 4);
      float4 f2 = *(const float4*)(rp + 16 + 8 * hi);
      float4 f3 = *(const float4*)(rp + 16 + 8 * hi + 4);
      a[0] = (_Float16)f0.x;  a[1] = (_Float16)f0.y;
      a[2] = (_Float16)f0.z;  a[3] = (_Float16)f0.w;
      a[4] = (_Float16)f1.x;  a[5] = (_Float16)f1.y;
      a[6] = (_Float16)f1.z;  a[7] = (_Float16)f1.w;
      a[8] = (_Float16)f2.x;  a[9] = (_Float16)f2.y;
      a[10] = (_Float16)f2.z; a[11] = (_Float16)f2.w;
      a[12] = (_Float16)f3.x; a[13] = (_Float16)f3.y;
      a[14] = (_Float16)f3.z; a[15] = (_Float16)f3.w;
    }
#pragma unroll
    for (int n = 0; n < 8; ++n) {
      v16h b = *(const v16h*)(Bs + ((k * 8 + n) * 32 + lane) * 16);
      acc[n] = __builtin_amdgcn_wmma_f32_16x16x32_f16(
          false, a, false, b, (short)0, acc[n], false, false);
    }
  }

  const float* bnp = bn + layer * LAT;
#pragma unroll
  for (int n = 0; n < 8; ++n) {
    float bias = bnp[n * 16 + m];
    int col = n * 16 + m;
#pragma unroll
    for (int v = 0; v < 8; ++v) {
      float x = acc[n][v] + bias;
      x = x > 0.0f ? x : 0.0f;
      int row = n0 + v + 8 * hi;
      if (write_out) h_out[(long)row * LAT + col] = x;
      h16[(long)row * LAT + col] = (_Float16)x;
    }
  }
}

extern "C" void kernel_launch(void* const* d_in, const int* in_sizes, int n_in,
                              void* d_out, int out_size, void* d_ws, size_t ws_size,
                              hipStream_t stream) {
  const float* nf     = (const float*)d_in[0];
  const float* ef     = (const float*)d_in[1];
  const int*   src    = (const int*)d_in[2];
  const int*   dst    = (const int*)d_in[3];
  const float* W_ne   = (const float*)d_in[4];
  const float* b_ne   = (const float*)d_in[5];
  const float* W_ee   = (const float*)d_in[6];
  const float* b_ee   = (const float*)d_in[7];
  const float* W_edge = (const float*)d_in[8];
  const float* b_edge = (const float*)d_in[9];
  const float* W_node = (const float*)d_in[10];
  const float* b_node = (const float*)d_in[11];

  float* h_out = (float*)d_out;                            // [N_NODES][128]
  float* e_out = (float*)d_out + (size_t)N_NODES * LAT;    // [N_EDGES][128]

  char* ws = (char*)d_ws;
  _Float16* h16 = (_Float16*)ws;  ws += (size_t)N_NODES * LAT * 2;   // 12.8 MB
  _Float16* e16 = (_Float16*)ws;  ws += (size_t)N_EDGES * LAT * 2;   // 153.6 MB
  float*    agg = (float*)ws;     ws += (size_t)N_NODES * LAT * 4;   // 25.6 MB
  _Float16* We16 = (_Float16*)ws; ws += (size_t)3 * 384 * LAT * 2;   // 288 KB
  _Float16* Wn16 = (_Float16*)ws;                                    // 192 KB

  pack_w<<<(147456 + 98304) / 256, 256, 0, stream>>>(W_edge, W_node, We16, Wn16);
  node_enc<<<(N_NODES * LAT) / 256, 256, 0, stream>>>(nf, W_ne, b_ne, h16);
  edge_enc<<<(N_EDGES * LAT) / 256, 256, 0, stream>>>(ef, W_ee, b_ee, e16);

  for (int l = 0; l < 3; ++l) {
    zero_f32<<<(N_NODES * LAT) / 256, 256, 0, stream>>>(agg, (long)N_NODES * LAT);
    edge_mlp<<<(EDGE_TILES + 7) / 8, 256, 0, stream>>>(
        h16, e16, e_out, src, dst, We16, b_edge, agg, l, (l == 2) ? 1 : 0);
    node_mlp<<<(NODE_TILES + 7) / 8, 256, 0, stream>>>(
        agg, h16, h_out, Wn16, b_node, l, (l == 2) ? 1 : 0);
  }
}
